// gunet_topk_84653805404429
// MI455X (gfx1250) — matrix operand order repaired and checked by reading, not compiled, not run on hardware
//
#include <hip/hip_runtime.h>

typedef __attribute__((ext_vector_type(2)))  float    v2f;
typedef __attribute__((ext_vector_type(8)))  float    v8f;
typedef __attribute__((ext_vector_type(16))) _Float16 v16h;
typedef __attribute__((ext_vector_type(8)))  _Float16 v8h;

// ---------------------------------------------------------------------------
// dinv[i] = rsqrt(1 + sum_j A[i,j])   (A has zero diagonal; Ah = A + I)
// ---------------------------------------------------------------------------
__global__ void rowsum_dinv_kernel(const float* __restrict__ A, int n,
                                   float* __restrict__ dinv) {
  __shared__ float sm[256];
  const int row = blockIdx.x, t = threadIdx.x;
  const float* ar = A + (size_t)row * n;
  float s = 0.f;
  for (int c = t; c < n; c += 256) s += ar[c];
  sm[t] = s; __syncthreads();
  for (int off = 128; off > 0; off >>= 1) {
    if (t < off) sm[t] += sm[t + off];
    __syncthreads();
  }
  if (t == 0) dinv[row] = rsqrtf(sm[0] + 1.0f);
}

// ---------------------------------------------------------------------------
// fp32 WMMA GEMM: C(MxN) = A(MxK) @ B(KxN), row-major. M%32==0, K%32==0,
// N arbitrary (guarded). 128 threads = 4 waves, 32x32 WG tile, 16x16 per wave.
// Tile staging is register-buffered: 16 unconditional global loads issue
// back-to-back (single wait), overlapping the previous tile's WMMA work.
// ---------------------------------------------------------------------------
__global__ __launch_bounds__(128) void gemm_f32_kernel(
    const float* __restrict__ A, int lda,
    const float* __restrict__ B, int ldb,
    float* __restrict__ C, int ldc, int M, int N, int K) {
  __shared__ float As[32][33];
  __shared__ float Bs[32][33];
  const int tid = threadIdx.x;
  const int wave = tid >> 5, lane = tid & 31;
  const int wm = wave >> 1, wn = wave & 1;
  const int g = lane >> 4, lm = lane & 15;
  const int m0 = blockIdx.y * 32;
  const int n0 = blockIdx.x * 32;
  // Each thread fills rows {tid>>5, +4, +8, ...}, column tid&31, of both tiles.
  const int lr = tid >> 5;       // base row within tile (0..3)
  const int lc = tid & 31;       // column within tile (0..31)
  const int bcol = n0 + lc;
  const bool colOk = (bcol < N);
  const int bcolC = colOk ? bcol : (N - 1);   // clamped -> loads always legal
  v8f acc = {};
  for (int kb = 0; kb < K; kb += 32) {
    float ar[8], br[8];
#pragma unroll
    for (int i = 0; i < 8; ++i) {
      const int r = lr + 4 * i;
      ar[i] = A[(size_t)(m0 + r) * lda + (kb + lc)];
      br[i] = B[(size_t)(kb + r) * ldb + bcolC];
    }
    __syncthreads();   // previous iteration done reading LDS
#pragma unroll
    for (int i = 0; i < 8; ++i) {
      const int r = lr + 4 * i;
      As[r][lc] = ar[i];
      Bs[r][lc] = colOk ? br[i] : 0.0f;
    }
    __syncthreads();
#pragma unroll
    for (int kk = 0; kk < 32; kk += 4) {
      const int klo = kk + 2 * g;
      v2f a, b;
      a.x = As[wm * 16 + lm][klo];
      a.y = As[wm * 16 + lm][klo + 1];
      b.x = Bs[klo][wn * 16 + lm];
      b.y = Bs[klo + 1][wn * 16 + lm];
      acc = __builtin_amdgcn_wmma_f32_16x16x4_f32(false, a, false, b,
                                                  (short)0, acc, false, false);
    }
  }
  const int gcol = n0 + wn * 16 + lm;
  if (gcol < N) {
#pragma unroll
    for (int r = 0; r < 8; ++r) {
      const int grow = m0 + wm * 16 + r + g * 8;
      C[(size_t)grow * ldc + gcol] = acc[r];
    }
  }
}

// ---------------------------------------------------------------------------
// f16 WMMA SYRK: S(kxk) = Gh(kxn) @ Gh^T, fp32 accum (exact for small ints),
// diagonal zeroed. One 16x16 tile per wave; 4 waves per block.
// ---------------------------------------------------------------------------
__global__ __launch_bounds__(128) void syrk_f16_kernel(
    const _Float16* __restrict__ G, int ldg, int kdim, int ndim,
    float* __restrict__ S, int lds_) {
  const int wave = threadIdx.x >> 5, lane = threadIdx.x & 31;
  const int m0 = (blockIdx.y * 4 + wave) * 16;
  const int n0 = blockIdx.x * 16;
  const int g = lane >> 4, lm = lane & 15;
  const _Float16* arow = G + (size_t)(m0 + lm) * ldg;
  const _Float16* brow = G + (size_t)(n0 + lm) * ldg;
  v8f acc = {};
#pragma unroll 2
  for (int t = 0; t < ndim; t += 32) {
    union { v16h v; v8h h[2]; } a;
    a.h[0] = *(const v8h*)(arow + t + g * 16);        // K = g*16 .. g*16+7
    a.h[1] = *(const v8h*)(arow + t + 8 + g * 16);    // K = 8+g*16 .. 15+g*16
    const v16h b = *(const v16h*)(brow + t + 16 * g); // K = 16g .. 16g+15
    acc = __builtin_amdgcn_wmma_f32_16x16x32_f16(false, a.v, false, b,
                                                 (short)0, acc, false, false);
  }
#pragma unroll
  for (int r = 0; r < 8; ++r) {
    const int row = m0 + r + g * 8;
    const int col = n0 + lm;
    S[(size_t)row * lds_ + col] = (row == col) ? 0.0f : acc[r];
  }
}

// ---------------------------------------------------------------------------
// fp32 WMMA SYRK (level 2, values may exceed f16-exact range)
// ---------------------------------------------------------------------------
__global__ __launch_bounds__(128) void syrk_f32_kernel(
    const float* __restrict__ G, int ldg, int kdim, int ndim,
    float* __restrict__ S, int lds_) {
  const int wave = threadIdx.x >> 5, lane = threadIdx.x & 31;
  const int m0 = (blockIdx.y * 4 + wave) * 16;
  const int n0 = blockIdx.x * 16;
  const int g = lane >> 4, lm = lane & 15;
  const float* arow = G + (size_t)(m0 + lm) * ldg;
  const float* brow = G + (size_t)(n0 + lm) * ldg;
  v8f acc = {};
#pragma unroll 4
  for (int t = 0; t < ndim; t += 4) {
    const v2f a = *(const v2f*)(arow + t + 2 * g);
    const v2f b = *(const v2f*)(brow + t + 2 * g);
    acc = __builtin_amdgcn_wmma_f32_16x16x4_f32(false, a, false, b,
                                                (short)0, acc, false, false);
  }
#pragma unroll
  for (int r = 0; r < 8; ++r) {
    const int row = m0 + r + g * 8;
    const int col = n0 + lm;
    S[(size_t)row * lds_ + col] = (row == col) ? 0.0f : acc[r];
  }
}

// ---------------------------------------------------------------------------
// score[i] = tanh( (x[i,:] . p) / ||p|| ),  H == blockDim.x == 256
// ---------------------------------------------------------------------------
__global__ void score_kernel(const float* __restrict__ x,
                             const float* __restrict__ p,
                             float* __restrict__ s, int H) {
  __shared__ float sdot[256];
  __shared__ float spp[256];
  const int row = blockIdx.x, t = threadIdx.x;
  const float xv = x[(size_t)row * H + t];
  const float pv = p[t];
  sdot[t] = xv * pv; spp[t] = pv * pv;
  __syncthreads();
  for (int off = 128; off > 0; off >>= 1) {
    if (t < off) { sdot[t] += sdot[t + off]; spp[t] += spp[t + off]; }
    __syncthreads();
  }
  if (t == 0) s[row] = tanhf(sdot[0] * rsqrtf(spp[0]));
}

// ---------------------------------------------------------------------------
// Single-block bitonic top-k. n is a power of two (<=4096). Sorts descending
// by score, ties -> lower index first (matches jax.lax.top_k).
// ---------------------------------------------------------------------------
__global__ __launch_bounds__(1024) void topk_kernel(
    const float* __restrict__ score, int n, int k,
    int* __restrict__ perm, float* __restrict__ vals) {
  __shared__ float skey[4096];
  __shared__ int sidx[4096];
  const int tid = threadIdx.x;
  for (int e = tid; e < n; e += 1024) { skey[e] = score[e]; sidx[e] = e; }
  __syncthreads();
  for (int size = 2; size <= n; size <<= 1) {
    for (int stride = size >> 1; stride > 0; stride >>= 1) {
      for (int i = tid; i < n; i += 1024) {
        const int j = i ^ stride;
        if (j > i) {
          const bool up = ((i & size) == 0);
          const float ki = skey[i], kj = skey[j];
          const int ii = sidx[i], ij = sidx[j];
          const bool iFirst = (ki > kj) || (ki == kj && ii < ij);
          if (up != iFirst) {
            skey[i] = kj; skey[j] = ki; sidx[i] = ij; sidx[j] = ii;
          }
        }
      }
      __syncthreads();
    }
  }
  for (int a = tid; a < k; a += 1024) { perm[a] = sidx[a]; vals[a] = skey[a]; }
}

// ---------------------------------------------------------------------------
// Gathers of G = (A + I)[perm, :]
// ---------------------------------------------------------------------------
__global__ void gather_gh_kernel(const float* __restrict__ A, int n,
                                 const int* __restrict__ perm,
                                 _Float16* __restrict__ Gh, int kdim) {
  const size_t idx = (size_t)blockIdx.x * 256 + threadIdx.x;
  if (idx >= (size_t)kdim * n) return;
  const int a = (int)(idx / n), t = (int)(idx % n);
  const int pa = perm[a];
  const float v = A[(size_t)pa * n + t] + ((t == pa) ? 1.0f : 0.0f);
  Gh[idx] = (_Float16)v;
}

__global__ void gather_gf_kernel(const float* __restrict__ A, int n,
                                 const int* __restrict__ perm,
                                 float* __restrict__ Gf, int kdim) {
  const size_t idx = (size_t)blockIdx.x * 256 + threadIdx.x;
  if (idx >= (size_t)kdim * n) return;
  const int a = (int)(idx / n), t = (int)(idx % n);
  const int pa = perm[a];
  Gf[idx] = A[(size_t)pa * n + t] + ((t == pa) ? 1.0f : 0.0f);
}

// ---------------------------------------------------------------------------
// Elementwise helpers
// ---------------------------------------------------------------------------
__global__ void pool_x_kernel(float* __restrict__ out, const float* __restrict__ x,
                              const int* __restrict__ perm,
                              const float* __restrict__ vals, int k, int H) {
  const int idx = blockIdx.x * 256 + threadIdx.x;
  if (idx >= k * H) return;
  const int a = idx / H, c = idx % H;
  out[idx] = x[(size_t)perm[a] * H + c] * vals[a];
}

__global__ void scale_rows_kernel(float* __restrict__ z,
                                  const float* __restrict__ dinv, int n, int cols) {
  const int idx = blockIdx.x * 256 + threadIdx.x;
  if (idx >= n * cols) return;
  z[idx] *= dinv[idx / cols];
}

// y holds A@z' ; out = dinv[i]*(y + z') + b[c], optional relu (in place on y)
__global__ void gcn_epilogue_kernel(float* __restrict__ y, const float* __restrict__ z,
                                    const float* __restrict__ dinv,
                                    const float* __restrict__ b, int n, int cols,
                                    int relu) {
  const int idx = blockIdx.x * 256 + threadIdx.x;
  if (idx >= n * cols) return;
  const int i = idx / cols, c = idx % cols;
  float v = dinv[i] * (y[idx] + z[idx]) + b[c];
  if (relu) v = fmaxf(v, 0.0f);
  y[idx] = v;
}

__global__ void copy_kernel(float* __restrict__ dst, const float* __restrict__ src,
                            int count) {
  const int idx = blockIdx.x * 256 + threadIdx.x;
  if (idx < count) dst[idx] = src[idx];
}

__global__ void scatter_add_kernel(float* __restrict__ dst,
                                   const float* __restrict__ xs,
                                   const int* __restrict__ perm, int k, int H) {
  const int idx = blockIdx.x * 256 + threadIdx.x;
  if (idx >= k * H) return;
  const int a = idx / H, c = idx % H;
  dst[(size_t)perm[a] * H + c] += xs[idx];  // perm indices are unique
}

// log_softmax over 16 classes, one wave per row (lanes 0..15 active)
__global__ void logsoftmax_kernel(const float* __restrict__ logits,
                                  float* __restrict__ out, int C) {
  const int row = blockIdx.x;
  const int lane = threadIdx.x;
  float v = (lane < C) ? logits[(size_t)row * C + lane] : -3.4e38f;
  float m = v;
  for (int off = 8; off > 0; off >>= 1) m = fmaxf(m, __shfl_xor(m, off, 16));
  float e = (lane < C) ? expf(v - m) : 0.0f;
  float s = e;
  for (int off = 8; off > 0; off >>= 1) s += __shfl_xor(s, off, 16);
  if (lane < C) out[(size_t)row * C + lane] = v - m - logf(s);
}

// ---------------------------------------------------------------------------
// Host orchestration
// ---------------------------------------------------------------------------
extern "C" void kernel_launch(void* const* d_in, const int* in_sizes, int n_in,
                              void* d_out, int out_size, void* d_ws, size_t ws_size,
                              hipStream_t stream) {
  (void)in_sizes; (void)n_in; (void)out_size; (void)ws_size;
  const float* X  = (const float*)d_in[0];   // 4096 x 128
  const float* A0 = (const float*)d_in[1];   // 4096 x 4096
  const float* Wd[4] = {(const float*)d_in[2], (const float*)d_in[3],
                        (const float*)d_in[4], (const float*)d_in[5]};
  const float* bd[4] = {(const float*)d_in[6], (const float*)d_in[7],
                        (const float*)d_in[8], (const float*)d_in[9]};
  const float* pw[3] = {(const float*)d_in[10], (const float*)d_in[11],
                        (const float*)d_in[12]};
  const float* Wu[3] = {(const float*)d_in[13], (const float*)d_in[14],
                        (const float*)d_in[15]};
  const float* bu[3] = {(const float*)d_in[16], (const float*)d_in[17],
                        (const float*)d_in[18]};
  float* out = (float*)d_out;

  char* base = (char*)d_ws;
  size_t off = 0;
  auto carve = [&](size_t bytes) -> char* {
    char* p = base + off;
    off += (bytes + 255) & ~(size_t)255;
    return p;
  };
  float* xs0  = (float*)carve((size_t)4096 * 256 * 4);
  float* xs1  = (float*)carve((size_t)2048 * 256 * 4);
  float* xs2  = (float*)carve((size_t)1024 * 256 * 4);
  float* x3   = (float*)carve((size_t)512 * 256 * 4);
  float* z    = (float*)carve((size_t)4096 * 256 * 4);
  float* xtmp = (float*)carve((size_t)4096 * 256 * 4);
  float* xup  = (float*)carve((size_t)4096 * 256 * 4);
  float* Alv1 = (float*)carve((size_t)2048 * 2048 * 4);
  float* Alv2 = (float*)carve((size_t)1024 * 1024 * 4);
  float* Alv3 = (float*)carve((size_t)512 * 512 * 4);
  _Float16* Gh = (_Float16*)carve((size_t)2048 * 4096 * 2);
  float* Gf = (float*)Gh;  // aliased: level-2 needs 512*1024*4 = 2MB < 16MB
  float* dinv0 = (float*)carve(4096 * 4);
  float* dinv1 = (float*)carve(2048 * 4);
  float* dinv2 = (float*)carve(1024 * 4);
  float* dinv3 = (float*)carve(512 * 4);
  float* score = (float*)carve(4096 * 4);
  float* vals  = (float*)carve(4096 * 4);
  int* perm0 = (int*)carve(2048 * 4);
  int* perm1 = (int*)carve(1024 * 4);
  int* perm2 = (int*)carve(512 * 4);

  auto blocks = [](size_t total) { return dim3((unsigned)((total + 255) / 256)); };
  auto gemm = [&](const float* A, int lda, const float* B, int ldb, float* Cm,
                  int ldc, int M, int N, int K) {
    dim3 g((N + 31) / 32, M / 32), b(128);
    gemm_f32_kernel<<<g, b, 0, stream>>>(A, lda, B, ldb, Cm, ldc, M, N, K);
  };
  // GCN: y = dinv⊙( A·(dinv⊙(x·W)) + dinv⊙(x·W) ) + b  [+relu]
  auto gcn = [&](const float* Alv, int n, const float* dinv, const float* xin,
                 const float* W, int fin, int fout, const float* bias, float* y,
                 int relu) {
    gemm(xin, fin, W, fout, z, fout, n, fout, fin);
    scale_rows_kernel<<<blocks((size_t)n * fout), 256, 0, stream>>>(z, dinv, n, fout);
    gemm(Alv, n, z, fout, y, fout, n, fout, n);
    gcn_epilogue_kernel<<<blocks((size_t)n * fout), 256, 0, stream>>>(y, z, dinv,
                                                                      bias, n, fout, relu);
  };

  // ---------------- Down level 0 ----------------
  rowsum_dinv_kernel<<<4096, 256, 0, stream>>>(A0, 4096, dinv0);
  gcn(A0, 4096, dinv0, X, Wd[0], 128, 256, bd[0], xs0, 1);

  // ---------------- Pool 0 (4096 -> 2048) ----------------
  score_kernel<<<4096, 256, 0, stream>>>(xs0, pw[0], score, 256);
  topk_kernel<<<1, 1024, 0, stream>>>(score, 4096, 2048, perm0, vals);
  pool_x_kernel<<<blocks((size_t)2048 * 256), 256, 0, stream>>>(xtmp, xs0, perm0, vals, 2048, 256);
  gather_gh_kernel<<<blocks((size_t)2048 * 4096), 256, 0, stream>>>(A0, 4096, perm0, Gh, 2048);
  syrk_f16_kernel<<<dim3(2048 / 16, 2048 / 64), 128, 0, stream>>>(Gh, 4096, 2048, 4096, Alv1, 2048);
  rowsum_dinv_kernel<<<2048, 256, 0, stream>>>(Alv1, 2048, dinv1);
  gcn(Alv1, 2048, dinv1, xtmp, Wd[1], 256, 256, bd[1], xs1, 1);

  // ---------------- Pool 1 (2048 -> 1024) ----------------
  score_kernel<<<2048, 256, 0, stream>>>(xs1, pw[1], score, 256);
  topk_kernel<<<1, 1024, 0, stream>>>(score, 2048, 1024, perm1, vals);
  pool_x_kernel<<<blocks((size_t)1024 * 256), 256, 0, stream>>>(xtmp, xs1, perm1, vals, 1024, 256);
  gather_gh_kernel<<<blocks((size_t)1024 * 2048), 256, 0, stream>>>(Alv1, 2048, perm1, Gh, 1024);
  syrk_f16_kernel<<<dim3(1024 / 16, 1024 / 64), 128, 0, stream>>>(Gh, 2048, 1024, 2048, Alv2, 1024);
  rowsum_dinv_kernel<<<1024, 256, 0, stream>>>(Alv2, 1024, dinv2);
  gcn(Alv2, 1024, dinv2, xtmp, Wd[2], 256, 256, bd[2], xs2, 1);

  // ---------------- Pool 2 (1024 -> 512) ----------------
  score_kernel<<<1024, 256, 0, stream>>>(xs2, pw[2], score, 256);
  topk_kernel<<<1, 1024, 0, stream>>>(score, 1024, 512, perm2, vals);
  pool_x_kernel<<<blocks((size_t)512 * 256), 256, 0, stream>>>(xtmp, xs2, perm2, vals, 512, 256);
  gather_gf_kernel<<<blocks((size_t)512 * 1024), 256, 0, stream>>>(Alv2, 1024, perm2, Gf, 512);
  syrk_f32_kernel<<<dim3(512 / 16, 512 / 64), 128, 0, stream>>>(Gf, 1024, 512, 1024, Alv3, 512);
  rowsum_dinv_kernel<<<512, 256, 0, stream>>>(Alv3, 512, dinv3);
  gcn(Alv3, 512, dinv3, xtmp, Wd[3], 256, 256, bd[3], x3, 1);

  // ---------------- Up j=2 (1024) ----------------
  copy_kernel<<<blocks((size_t)1024 * 256), 256, 0, stream>>>(xtmp, xs2, 1024 * 256);
  scatter_add_kernel<<<blocks((size_t)512 * 256), 256, 0, stream>>>(xtmp, x3, perm2, 512, 256);
  gcn(Alv2, 1024, dinv2, xtmp, Wu[0], 256, 256, bu[0], xup, 1);

  // ---------------- Up j=1 (2048) ----------------
  copy_kernel<<<blocks((size_t)2048 * 256), 256, 0, stream>>>(xtmp, xs1, 2048 * 256);
  scatter_add_kernel<<<blocks((size_t)1024 * 256), 256, 0, stream>>>(xtmp, xup, perm1, 1024, 256);
  gcn(Alv1, 2048, dinv1, xtmp, Wu[1], 256, 256, bu[1], xup, 1);

  // ---------------- Up j=0 (4096), 256 -> 16, no relu ----------------
  copy_kernel<<<blocks((size_t)4096 * 256), 256, 0, stream>>>(xtmp, xs0, 4096 * 256);
  scatter_add_kernel<<<blocks((size_t)2048 * 256), 256, 0, stream>>>(xtmp, xup, perm0, 2048, 256);
  gcn(A0, 4096, dinv0, xtmp, Wu[2], 256, 16, bu[2], xup, 0);

  logsoftmax_kernel<<<4096, 32, 0, stream>>>(xup, out, 16);
}
